// GAT_57415122813718
// MI455X (gfx1250) — matrix operand order repaired and checked by reading, not compiled, not run on hardware
//
#include <hip/hip_runtime.h>

// ---------------------------------------------------------------------------
// Types (clang ext_vector: trivially copyable, exact WMMA operand types)
// ---------------------------------------------------------------------------
typedef __attribute__((ext_vector_type(16))) __bf16    v16bf;
typedef __attribute__((ext_vector_type(8)))  float     v8f;
typedef __attribute__((ext_vector_type(4)))  float     f32x4;
typedef __attribute__((ext_vector_type(4)))  unsigned  u32x4;
typedef __attribute__((ext_vector_type(8)))  int       i32x8;
typedef __attribute__((ext_vector_type(4)))  int       i32x4;

union FragCast { u32x4 u[2]; v16bf v; };

#define GEMM_BM 128
#define GEMM_BN 64
#define GEMM_BK 32

#if __has_builtin(__builtin_amdgcn_tensor_load_to_lds) && \
    __has_builtin(__builtin_amdgcn_s_wait_tensorcnt)
#define USE_TDM 1
#else
#define USE_TDM 0
#endif

#if USE_TDM
// ---------------------------------------------------------------------------
// Issue one 2D TDM tile load (bf16 elements): LDS tile is tile_w x tile_h,
// rows contiguous (row stride = tile_w*2 bytes). OOB rows/cols read as zero.
// D# layout per CDNA5 ISA. 6-arg builtin
// (clang-23 lane): (v4u g0, v8i g1, v4i g2, v4i g3, v8i pad, i32 cpol).
// ---------------------------------------------------------------------------
__device__ __forceinline__ void tdm_tile_load(
    const __bf16* lds_dst, const __bf16* g_src,
    int rem_w, int rem_h, int tile_w, int tile_h, int stride_elems)
{
  unsigned long long ga = (unsigned long long)(size_t)g_src;
  unsigned lds = (unsigned)(size_t)lds_dst;   // low 32 bits of LDS aperture = offset
  u32x4 g0;
  g0[0] = 1u;                                          // count=1, user descriptor
  g0[1] = lds;                                         // lds_addr
  g0[2] = (unsigned)ga;                                // global_addr[31:0]
  g0[3] = (unsigned)((ga >> 32) & 0x01ffffffull)       // global_addr[56:32]
        | (2u << 30);                                  // type = 2 ("image")
  i32x8 g1;
  g1[0] = (int)(1u << 16);                             // data_size=1 (2 bytes); no pad
  g1[1] = (int)(((unsigned)rem_w & 0xffffu) << 16);    // tensor_dim0[15:0]
  g1[2] = (int)((((unsigned)rem_w >> 16) & 0xffffu)    // tensor_dim0[31:16]
        | (((unsigned)rem_h & 0xffffu) << 16));        // tensor_dim1[15:0]
  g1[3] = (int)((((unsigned)rem_h >> 16) & 0xffffu)    // tensor_dim1[31:16]
        | (((unsigned)tile_w & 0xffffu) << 16));       // tile_dim0
  g1[4] = (int)((unsigned)tile_h & 0xffffu);           // tile_dim1 (tile_dim2=0)
  g1[5] = (int)stride_elems;                           // tensor_dim0_stride[31:0]
  g1[6] = 0;
  g1[7] = 0;
  i32x4 z4 = {};
  i32x8 z8 = {};
  __builtin_amdgcn_tensor_load_to_lds(g0, g1, z4, z4, z8, 0);
}
#endif

// ---------------------------------------------------------------------------
// Pre-split f32 -> bf16 hi/lo planes (elementwise)
// ---------------------------------------------------------------------------
__global__ void gat_split_bf16(const float* __restrict__ in,
                               __bf16* __restrict__ hi, __bf16* __restrict__ lo,
                               size_t total4)
{
  size_t i = (size_t)blockIdx.x * blockDim.x + threadIdx.x;
  if (i >= total4) return;
  size_t base = i * 4;
  f32x4 v = *(const f32x4*)(in + base);
#pragma unroll
  for (int j = 0; j < 4; ++j) {
    float f = v[j];
    __bf16 h = (__bf16)f;
    hi[base + j] = h;
    lo[base + j] = (__bf16)(f - (float)h);
  }
}

// Split + transpose W[Fin,Kout] -> planes Wt[Kout,Fin] (K-contiguous rows)
__global__ void gat_split_transpose_w(const float* __restrict__ Wm,
                                      __bf16* __restrict__ hi, __bf16* __restrict__ lo,
                                      int Fin, int Kout)
{
  int idx = blockIdx.x * blockDim.x + threadIdx.x;
  if (idx >= Fin * Kout) return;
  int k = idx / Kout, n = idx - k * Kout;
  float f = Wm[idx];
  __bf16 h = (__bf16)f;
  size_t o = (size_t)n * Fin + k;
  hi[o] = h;
  lo[o] = (__bf16)(f - (float)h);
}

// ---------------------------------------------------------------------------
// GEMM: O[N,Kout] = X[N,Fin] * W[Fin,Kout], pre-split bf16 hi/lo operands.
// block = 256 threads (8 waves, 4x2), wave tile 32x32 (2x2 wmma tiles).
// TDM stages tiles into double-buffered LDS; compute overlaps next DMA.
// ---------------------------------------------------------------------------
__global__ __launch_bounds__(256) void gat_gemm_wmma(
    const __bf16* __restrict__ Xhi, const __bf16* __restrict__ Xlo,
    const __bf16* __restrict__ Whi, const __bf16* __restrict__ Wlo,  // [Kout][Fin]
    float* __restrict__ O, int N, int Fin, int Kout)
{
  __shared__ __bf16 As[2][2][GEMM_BM * GEMM_BK];   // [buf][hi/lo]
  __shared__ __bf16 Bs[2][2][GEMM_BN * GEMM_BK];

  const int tid  = threadIdx.x;
  const int lane = tid & 31;
  const int wv   = tid >> 5;
  const int wm   = wv & 3;       // wave row  (M)
  const int wn   = wv >> 2;      // wave col  (N)
  const int bM   = blockIdx.x * GEMM_BM;
  const int bN   = blockIdx.y * GEMM_BN;
  const int lh   = lane & 15;
  const int hi16 = lane >> 4;    // 0 or 1

  v8f acc[2][2] = {};
  const int nk = Fin >> 5;

#if USE_TDM
  if (wv == 0) {
    tdm_tile_load(&As[0][0][0], Xhi + (size_t)bM * Fin, Fin, N - bM,    GEMM_BK, GEMM_BM, Fin);
    tdm_tile_load(&As[0][1][0], Xlo + (size_t)bM * Fin, Fin, N - bM,    GEMM_BK, GEMM_BM, Fin);
    tdm_tile_load(&Bs[0][0][0], Whi + (size_t)bN * Fin, Fin, Kout - bN, GEMM_BK, GEMM_BN, Fin);
    tdm_tile_load(&Bs[0][1][0], Wlo + (size_t)bN * Fin, Fin, Kout - bN, GEMM_BK, GEMM_BN, Fin);
  }
#endif

  for (int kt = 0; kt < nk; ++kt) {
#if USE_TDM
    const int buf = kt & 1;
    __builtin_amdgcn_s_wait_tensorcnt(0);   // nonzero only for the issuing wave
    __syncthreads();                         // everyone sees buf[kt] ready
    if (wv == 0 && kt + 1 < nk) {            // prefetch next tile while computing
      const int nb = buf ^ 1;
      const int k1 = (kt + 1) << 5;
      tdm_tile_load(&As[nb][0][0], Xhi + (size_t)bM * Fin + k1, Fin - k1, N - bM,    GEMM_BK, GEMM_BM, Fin);
      tdm_tile_load(&As[nb][1][0], Xlo + (size_t)bM * Fin + k1, Fin - k1, N - bM,    GEMM_BK, GEMM_BM, Fin);
      tdm_tile_load(&Bs[nb][0][0], Whi + (size_t)bN * Fin + k1, Fin - k1, Kout - bN, GEMM_BK, GEMM_BN, Fin);
      tdm_tile_load(&Bs[nb][1][0], Wlo + (size_t)bN * Fin + k1, Fin - k1, Kout - bN, GEMM_BK, GEMM_BN, Fin);
    }
#else
    const int buf = 0;
    const int k0 = kt << 5;
    // cooperative staging: A planes 512 uint4 each, B planes 256 uint4 each
#pragma unroll
    for (int pl = 0; pl < 2; ++pl) {
      const __bf16* srcA = pl ? Xlo : Xhi;
#pragma unroll
      for (int i = 0; i < 2; ++i) {
        int idx = tid + i * 256;       // 0..511
        int r = idx >> 2;              // row 0..127
        int c = (idx & 3) << 3;        // 0,8,16,24
        u32x4 v = {};
        int gr = bM + r;
        if (gr < N) v = *(const u32x4*)(srcA + (size_t)gr * Fin + k0 + c);
        *(u32x4*)(&As[0][pl][r * GEMM_BK + c]) = v;
      }
      const __bf16* srcB = pl ? Wlo : Whi;
      int r = tid >> 2;
      int c = (tid & 3) << 3;
      u32x4 v = *(const u32x4*)(srcB + (size_t)(bN + r) * Fin + k0 + c);
      *(u32x4*)(&Bs[0][pl][r * GEMM_BK + c]) = v;
    }
    __syncthreads();
#endif

    // ---- fragment loads per documented CDNA5 VGPR layouts ----
    v16bf afh[2], afl[2], bfh[2], bfl[2];
#pragma unroll
    for (int mt = 0; mt < 2; ++mt) {
      int m  = wm * 32 + mt * 16 + lh;
      int kb = hi16 ? 8 : 0;
      FragCast fc;
      const __bf16* p = &As[buf][0][m * GEMM_BK + kb];
      fc.u[0] = *(const u32x4*)(p);
      fc.u[1] = *(const u32x4*)(p + 16);
      afh[mt] = fc.v;
      const __bf16* q = &As[buf][1][m * GEMM_BK + kb];
      fc.u[0] = *(const u32x4*)(q);
      fc.u[1] = *(const u32x4*)(q + 16);
      afl[mt] = fc.v;
    }
#pragma unroll
    for (int nt = 0; nt < 2; ++nt) {
      int n  = wn * 32 + nt * 16 + lh;
      int kb = hi16 ? 8 : 0;
      FragCast fc;
      const __bf16* p = &Bs[buf][0][n * GEMM_BK + kb];
      fc.u[0] = *(const u32x4*)(p);
      fc.u[1] = *(const u32x4*)(p + 16);
      bfh[nt] = fc.v;
      const __bf16* q = &Bs[buf][1][n * GEMM_BK + kb];
      fc.u[0] = *(const u32x4*)(q);
      fc.u[1] = *(const u32x4*)(q + 16);
      bfl[nt] = fc.v;
    }

    // ---- 3-term split product: hi*hi + lo*hi + hi*lo (~f32 accuracy) ----
#pragma unroll
    for (int mt = 0; mt < 2; ++mt)
#pragma unroll
      for (int nt = 0; nt < 2; ++nt) {
        acc[mt][nt] = __builtin_amdgcn_wmma_f32_16x16x32_bf16(
            false, afh[mt], false, bfh[nt], (short)0, acc[mt][nt], false, false);
        acc[mt][nt] = __builtin_amdgcn_wmma_f32_16x16x32_bf16(
            false, afl[mt], false, bfh[nt], (short)0, acc[mt][nt], false, false);
        acc[mt][nt] = __builtin_amdgcn_wmma_f32_16x16x32_bf16(
            false, afh[mt], false, bfl[nt], (short)0, acc[mt][nt], false, false);
      }
#if !USE_TDM
    __syncthreads();
#endif
  }

  // ---- store C (VGPR v: lanes0-15 -> M=v, lanes16-31 -> M=v+8; N = lane&15) ----
#pragma unroll
  for (int mt = 0; mt < 2; ++mt) {
#pragma unroll
    for (int nt = 0; nt < 2; ++nt) {
      int col = bN + wn * 32 + nt * 16 + lh;
#pragma unroll
      for (int r8 = 0; r8 < 8; ++r8) {
        int row = bM + wm * 32 + mt * 16 + r8 + (hi16 ? 8 : 0);
        if (row < N) O[(size_t)row * Kout + col] = acc[mt][nt][r8];
      }
    }
  }
}

// ---------------------------------------------------------------------------
// Attention logits: al_s[n,h] = sum_c h[n,h,c]*a_src[h,c]; one wave per (n,h)
// ---------------------------------------------------------------------------
__global__ void gat_attn_logits(const float* __restrict__ hfeat,
                                const float* __restrict__ a_src,
                                const float* __restrict__ a_dst,
                                float* __restrict__ als, float* __restrict__ ald,
                                int N, int H, int C)
{
  int gwave = (int)((blockIdx.x * blockDim.x + threadIdx.x) >> 5);
  int lane  = threadIdx.x & 31;
  if (gwave >= N * H) return;
  int n  = gwave / H;
  int hd = gwave - n * H;
  const float* hp = hfeat + (size_t)n * H * C + (size_t)hd * C;
  const float* as = a_src + hd * C;
  const float* ad = a_dst + hd * C;
  float ss = 0.0f, sd = 0.0f;
  for (int c = lane; c < C; c += 32) {
    float v = hp[c];
    ss += v * as[c];
    sd += v * ad[c];
  }
  for (int off = 16; off > 0; off >>= 1) {
    ss += __shfl_xor(ss, off, 32);
    sd += __shfl_xor(sd, off, 32);
  }
  if (lane == 0) { als[gwave] = ss; ald[gwave] = sd; }
}

// ---------------------------------------------------------------------------
// Monotonic float<->uint map so segment-max can use hardware atomicMax(u32)
// ---------------------------------------------------------------------------
__device__ __forceinline__ unsigned f2ord(float f) {
  unsigned u = __float_as_uint(f);
  return (u & 0x80000000u) ? ~u : (u | 0x80000000u);
}
__device__ __forceinline__ float ord2f(unsigned o) {
  unsigned u = (o & 0x80000000u) ? (o & 0x7fffffffu) : ~o;
  return __uint_as_float(u);
}

__global__ void gat_edge_max(const int* __restrict__ ei,
                             const float* __restrict__ als, const float* __restrict__ ald,
                             unsigned* __restrict__ mord, int E, int Nn, int H)
{
  int idx = blockIdx.x * blockDim.x + threadIdx.x;
  int tot = (E + Nn) * H;
  if (idx >= tot) return;
  int e = idx / H, hd = idx - e * H;
  int s = (e < E) ? ei[e]     : (e - E);   // self-loops appended after real edges
  int d = (e < E) ? ei[E + e] : (e - E);
  float v = als[s * H + hd] + ald[d * H + hd];
  v = (v > 0.0f) ? v : v * 0.2f;           // leaky_relu(0.2)
  atomicMax(&mord[d * H + hd], f2ord(v));
}

__global__ void gat_edge_exp(const int* __restrict__ ei,
                             const float* __restrict__ als, const float* __restrict__ ald,
                             const unsigned* __restrict__ mord,
                             float* __restrict__ denom, float* __restrict__ exbuf,
                             int E, int Nn, int H)
{
  int idx = blockIdx.x * blockDim.x + threadIdx.x;
  int tot = (E + Nn) * H;
  if (idx >= tot) return;
  int e = idx / H, hd = idx - e * H;
  int s = (e < E) ? ei[e]     : (e - E);
  int d = (e < E) ? ei[E + e] : (e - E);
  float v = als[s * H + hd] + ald[d * H + hd];
  v = (v > 0.0f) ? v : v * 0.2f;
  float m  = ord2f(mord[d * H + hd]);
  float ex = __expf(v - m);
  exbuf[idx] = ex;
  atomicAdd(&denom[d * H + hd], ex);
}

// ---------------------------------------------------------------------------
// Aggregation: agg[dst,h,c] += (ex/denom[dst,h]) * hfeat[src,h,c]
// one block per edge; float4 gathers, f32 atomic adds (L2-resident)
// ---------------------------------------------------------------------------
__global__ void gat_edge_agg(const int* __restrict__ ei,
                             const float* __restrict__ hfeat,
                             const float* __restrict__ exbuf,
                             const float* __restrict__ denom,
                             float* __restrict__ agg,
                             int E, int Nn, int H, int C)
{
  int e = blockIdx.x;
  int s = (e < E) ? ei[e]     : (e - E);
  int d = (e < E) ? ei[E + e] : (e - E);
  int HC = H * C;
  const f32x4* hp = (const f32x4*)(hfeat + (size_t)s * HC);
  for (int c4 = threadIdx.x; c4 < (HC >> 2); c4 += blockDim.x) {
    int hd = (c4 << 2) / C;
    float alpha = exbuf[(size_t)e * H + hd] / denom[d * H + hd];
    f32x4 v = hp[c4];
    float* ob = agg + (size_t)d * HC + (c4 << 2);
    atomicAdd(ob + 0, alpha * v[0]);
    atomicAdd(ob + 1, alpha * v[1]);
    atomicAdd(ob + 2, alpha * v[2]);
    atomicAdd(ob + 3, alpha * v[3]);
  }
}

// ---------------------------------------------------------------------------
// Epilogue: out = act(agg + bias); mode 0 = relu, 1 = tanh
// ---------------------------------------------------------------------------
__global__ void gat_epilogue(const float* __restrict__ agg, const float* __restrict__ b,
                             float* __restrict__ out, int total, int HC, int mode)
{
  int i = blockIdx.x * blockDim.x + threadIdx.x;
  if (i >= total) return;
  float v = agg[i] + b[i % HC];
  out[i] = mode ? tanhf(v) : fmaxf(v, 0.0f);
}

// ---------------------------------------------------------------------------
// Host orchestration
// ---------------------------------------------------------------------------
static void run_layer(const float* xin, int N, int E, int Fin, int H, int C,
                      const float* W, const float* as, const float* ad, const float* b,
                      const int* ei,
                      __bf16* xhi, __bf16* xlo, __bf16* whi, __bf16* wlo,
                      float* buf_h, float* buf_agg, float* als, float* ald,
                      unsigned* mord, float* denom, float* exbuf,
                      float* out, int act, hipStream_t stream)
{
  const int HC   = H * C;
  const int Etot = E + N;

  // pre-split operands to bf16 hi/lo planes (W also transposed to [HC][Fin])
  size_t xtot4 = ((size_t)N * Fin) >> 2;
  gat_split_bf16<<<(unsigned)((xtot4 + 255) / 256), 256, 0, stream>>>(xin, xhi, xlo, xtot4);
  int wtot = Fin * HC;
  gat_split_transpose_w<<<(wtot + 255) / 256, 256, 0, stream>>>(W, whi, wlo, Fin, HC);

  dim3 gGemm((N + GEMM_BM - 1) / GEMM_BM, HC / GEMM_BN);
  gat_gemm_wmma<<<gGemm, 256, 0, stream>>>(xhi, xlo, whi, wlo, buf_h, N, Fin, HC);

  (void)hipMemsetAsync(mord,  0, (size_t)N * H * sizeof(unsigned), stream);
  (void)hipMemsetAsync(denom, 0, (size_t)N * H * sizeof(float), stream);
  (void)hipMemsetAsync(buf_agg, 0, (size_t)N * HC * sizeof(float), stream);

  int nw = N * H;  // one wave per (node, head)
  gat_attn_logits<<<(nw + 7) / 8, 256, 0, stream>>>(buf_h, as, ad, als, ald, N, H, C);

  int tot = Etot * H;
  gat_edge_max<<<(tot + 255) / 256, 256, 0, stream>>>(ei, als, ald, mord, E, N, H);
  gat_edge_exp<<<(tot + 255) / 256, 256, 0, stream>>>(ei, als, ald, mord, denom, exbuf, E, N, H);

  int bs = HC / 4;
  if (bs > 256) bs = 256;
  if (bs < 32)  bs = 32;
  gat_edge_agg<<<Etot, bs, 0, stream>>>(ei, buf_h, exbuf, denom, buf_agg, E, N, H, C);

  int totn = N * HC;
  gat_epilogue<<<(totn + 255) / 256, 256, 0, stream>>>(buf_agg, b, out, totn, HC, act);
}

static inline size_t alignup(size_t x) { return (x + 255) & ~(size_t)255; }

extern "C" void kernel_launch(void* const* d_in, const int* in_sizes, int n_in,
                              void* d_out, int out_size, void* d_ws, size_t ws_size,
                              hipStream_t stream)
{
  const float* x   = (const float*)d_in[0];
  const int*   ei  = (const int*)d_in[1];
  const float* W1  = (const float*)d_in[2];
  const float* a1s = (const float*)d_in[3];
  const float* a1d = (const float*)d_in[4];
  const float* b1  = (const float*)d_in[5];
  const float* W2  = (const float*)d_in[6];
  const float* a2s = (const float*)d_in[7];
  const float* a2d = (const float*)d_in[8];
  const float* b2  = (const float*)d_in[9];
  const float* W3  = (const float*)d_in[10];
  const float* a3s = (const float*)d_in[11];
  const float* a3d = (const float*)d_in[12];
  const float* b3  = (const float*)d_in[13];

  const int N = in_sizes[0] / 768;   // 20000
  const int E = in_sizes[1] / 2;     // 320000
  const int Etot = E + N;
  const int HMAX = 4;

  // workspace carve-up
  char* ws = (char*)d_ws;
  size_t off = 0;
  float* buf_h   = (float*)(ws + off); off = alignup(off + (size_t)N * 1024 * 4);
  float* buf_agg = (float*)(ws + off); off = alignup(off + (size_t)N * 1024 * 4);
  float* featA   = (float*)(ws + off); off = alignup(off + (size_t)N * 1024 * 4);
  float* featB   = (float*)(ws + off); off = alignup(off + (size_t)N * 512  * 4);
  __bf16* xhi    = (__bf16*)(ws + off); off = alignup(off + (size_t)N * 1024 * 2);
  __bf16* xlo    = (__bf16*)(ws + off); off = alignup(off + (size_t)N * 1024 * 2);
  __bf16* whi    = (__bf16*)(ws + off); off = alignup(off + (size_t)1024 * 1024 * 2);
  __bf16* wlo    = (__bf16*)(ws + off); off = alignup(off + (size_t)1024 * 1024 * 2);
  float* als     = (float*)(ws + off); off = alignup(off + (size_t)N * HMAX * 4);
  float* ald     = (float*)(ws + off); off = alignup(off + (size_t)N * HMAX * 4);
  unsigned* mord = (unsigned*)(ws + off); off = alignup(off + (size_t)N * HMAX * 4);
  float* denom   = (float*)(ws + off); off = alignup(off + (size_t)N * HMAX * 4);
  float* exbuf   = (float*)(ws + off); off = alignup(off + (size_t)Etot * HMAX * 4);
  (void)ws_size; (void)n_in; (void)out_size;

  // layer 1: 768 -> 4x256 (concat 1024), relu
  run_layer(x,     N, E,  768, 4, 256, W1, a1s, a1d, b1, ei, xhi, xlo, whi, wlo,
            buf_h, buf_agg, als, ald, mord, denom, exbuf, featA, /*act=*/0, stream);
  // layer 2: 1024 -> 4x128 (concat 512), relu
  run_layer(featA, N, E, 1024, 4, 128, W2, a2s, a2d, b2, ei, xhi, xlo, whi, wlo,
            buf_h, buf_agg, als, ald, mord, denom, exbuf, featB, /*act=*/0, stream);
  // layer 3: 512 -> 1x64 (mean over H=1 is identity), tanh
  run_layer(featB, N, E,  512, 1,  64, W3, a3s, a3d, b3, ei, xhi, xlo, whi, wlo,
            buf_h, buf_agg, als, ald, mord, denom, exbuf, (float*)d_out, /*act=*/1, stream);
}
